// SAB_82643760710412
// MI455X (gfx1250) — compile-verified, matrix operand order checked by reading, not run on hardware
//
#include <hip/hip_runtime.h>
#include <math.h>

// ---------------------------------------------------------------------------
// MI455X (gfx1250, wave32) fused pipeline for the linear-attention block.
//   K0: init + f16 weights + wl row-sum constants
//   K1: K,V proj (WMMA) -> S[b,h] = sum_n K*V
//   K2: Q,V proj (WMMA) -> Wpre regs -> LDS -> Wpre @ Wl^T (WMMA) -> G=d_out
//       + 5 per-batch scalars (both LayerNorms reduce to closed form)
//   K3: out = tanh(a*G + b[o])   (v_tanh_f32 when available)
// Workspace need: ~167 KB.
// ---------------------------------------------------------------------------

typedef __attribute__((ext_vector_type(16))) _Float16 v16h;
typedef __attribute__((ext_vector_type(8)))  float    v8f;

#define BB   64
#define NN   4096
#define DIM  128
#define RPB  128          // rows handled per workgroup
#define LDST 136          // LDS row stride in halfs (bank spread)
#define LN_CNT (NN * DIM) // 524288 elements per batch LayerNorm
#define EPS  1e-5f

// workspace byte offsets
#define WS_WQ16   0u
#define WS_WK16   32768u
#define WS_WV16   65536u
#define WS_WL16   98304u
#define WS_WLSUM  131072u   // 128 floats wl_s[o] + [128]=sum(wl_s) + [129]=sum(wl_s^2)
#define WS_S      132096u   // 64*128 floats
#define WS_ST     164864u   // 64*8 floats: {sumW, sumW2, sumG, sumG2, sumGw} per batch

#if __has_builtin(__builtin_amdgcn_tanhf)
__device__ __forceinline__ float fast_tanh(float x) { return __builtin_amdgcn_tanhf(x); }
#elif __has_builtin(__builtin_amdgcn_tanh_f32)
__device__ __forceinline__ float fast_tanh(float x) { return __builtin_amdgcn_tanh_f32(x); }
#else
__device__ __forceinline__ float fast_tanh(float x) { return tanhf(x); }
#endif

// Load a 16x32 B-fragment from a row-major [out=128][in=128] f16 weight.
__device__ __forceinline__ v16h frag_from_w(const _Float16* w, int n, int k0, int klo) {
    v16h r;
    const _Float16* p = w + n * DIM + k0 + klo;
#pragma unroll
    for (int i = 0; i < 8; ++i) { r[i] = p[i]; r[8 + i] = p[16 + i]; }
    return r;
}

// ---------------------------------------------------------------------------
__global__ void __launch_bounds__(256)
k0_init(const float* __restrict__ Wq, const float* __restrict__ Wk,
        const float* __restrict__ Wv, const float* __restrict__ Wl,
        _Float16* wq16, _Float16* wk16, _Float16* wv16, _Float16* wl16,
        float* wlsum, float* S, float* st) {
    int t = threadIdx.x;
    for (int i = t; i < DIM * DIM; i += 256) {
        wq16[i] = (_Float16)Wq[i];
        wk16[i] = (_Float16)Wk[i];
        wv16[i] = (_Float16)Wv[i];
        wl16[i] = (_Float16)Wl[i];
    }
    for (int i = t; i < BB * DIM; i += 256) S[i] = 0.f;
    for (int i = t; i < 8 * BB; i += 256) st[i] = 0.f;
    if (t < DIM) {
        float s = 0.f;
        for (int h = 0; h < DIM; ++h) s += Wl[t * DIM + h];
        wlsum[t] = s;
    }
    __syncthreads();
    if (t == 0) {
        float s1 = 0.f, s2 = 0.f;
        for (int o = 0; o < DIM; ++o) { float v = wlsum[o]; s1 += v; s2 += v * v; }
        wlsum[DIM]     = s1;   // sum_o wl_s[o]
        wlsum[DIM + 1] = s2;   // sum_o wl_s[o]^2
    }
}

// ---------------------------------------------------------------------------
// K1: K = X@Wk^T, V = X@Wv^T (WMMA), S[b,h] += sum_rows K*V
__global__ void __launch_bounds__(256)
k1_kv_s(const float* __restrict__ X, const _Float16* __restrict__ wk16,
        const _Float16* __restrict__ wv16, float* S) {
    __shared__ _Float16 xs[RPB * LDST];
    __shared__ float sS[DIM];
    const int b  = blockIdx.y;
    const int r0 = blockIdx.x * RPB;
    const int tid = threadIdx.x;

    const float* xp = X + ((size_t)b * NN + r0) * DIM;
    for (int i = tid; i < RPB * DIM; i += 256) {
        int rr = i >> 7, cc = i & 127;
        xs[rr * LDST + cc] = (_Float16)xp[rr * DIM + cc];
    }
    if (tid < DIM) sS[tid] = 0.f;
    __syncthreads();

    const int w = tid >> 5, lane = tid & 31;
    const int half = lane >> 4, nlo = lane & 15, klo = half * 8;
    const int nglob = w * 16 + nlo;

    v16h bk[4], bv[4];
#pragma unroll
    for (int ks = 0; ks < 4; ++ks) {
        bk[ks] = frag_from_w(wk16, nglob, ks * 32, klo);
        bv[ks] = frag_from_w(wv16, nglob, ks * 32, klo);
    }

    float part = 0.f;
#pragma unroll
    for (int mt = 0; mt < 8; ++mt) {
        v8f ck = {}; v8f cv = {};
#pragma unroll
        for (int ks = 0; ks < 4; ++ks) {
            v16h a;
            const int base = (mt * 16 + nlo) * LDST + ks * 32 + klo;
#pragma unroll
            for (int i = 0; i < 8; ++i) { a[i] = xs[base + i]; a[8 + i] = xs[base + 16 + i]; }
            ck = __builtin_amdgcn_wmma_f32_16x16x32_f16(false, a, false, bk[ks], (short)0, ck, false, false);
            cv = __builtin_amdgcn_wmma_f32_16x16x32_f16(false, a, false, bv[ks], (short)0, cv, false, false);
        }
#pragma unroll
        for (int r = 0; r < 8; ++r) part += ck[r] * cv[r];
    }
    atomicAdd(&sS[nglob], part);
    __syncthreads();
    if (tid < DIM) atomicAdd(&S[b * DIM + tid], sS[tid]);
}

// ---------------------------------------------------------------------------
// K2 (fused): Wpre = Q*(S/N)+V in registers -> LDS -> G = Wpre@Wl^T -> d_out.
// Accumulates per-batch {sumW, sumW2, sumG, sumG2, sumGw}.
__global__ void __launch_bounds__(256)
k2_fused(const float* __restrict__ X, const _Float16* __restrict__ wq16,
         const _Float16* __restrict__ wv16, const _Float16* __restrict__ wl16,
         const float* __restrict__ S, const float* __restrict__ wlsum,
         float* G, float* st) {
    __shared__ _Float16 xs[RPB * LDST];    // X tile (f16)
    __shared__ _Float16 wsh[RPB * LDST];   // Wpre tile (f16) for stage 2
    __shared__ float red[8];
    const int b  = blockIdx.y;
    const int r0 = blockIdx.x * RPB;
    const int tid = threadIdx.x;

    const float* xp = X + ((size_t)b * NN + r0) * DIM;
    for (int i = tid; i < RPB * DIM; i += 256) {
        int rr = i >> 7, cc = i & 127;
        xs[rr * LDST + cc] = (_Float16)xp[rr * DIM + cc];
    }
    if (tid < 8) red[tid] = 0.f;
    __syncthreads();

    const int w = tid >> 5, lane = tid & 31;
    const int half = lane >> 4, nlo = lane & 15, klo = half * 8;
    const int nglob = w * 16 + nlo;
    const float sscale = S[b * DIM + nglob] * (1.0f / NN);

    // ---- stage 1: Wpre = Q*(S/N) + V ----
    v16h bq[4], bv[4];
#pragma unroll
    for (int ks = 0; ks < 4; ++ks) {
        bq[ks] = frag_from_w(wq16, nglob, ks * 32, klo);
        bv[ks] = frag_from_w(wv16, nglob, ks * 32, klo);
    }
    float sumW = 0.f, sumW2 = 0.f;
#pragma unroll
    for (int mt = 0; mt < 8; ++mt) {
        v8f cq = {}; v8f cv = {};
#pragma unroll
        for (int ks = 0; ks < 4; ++ks) {
            v16h a;
            const int base = (mt * 16 + nlo) * LDST + ks * 32 + klo;
#pragma unroll
            for (int i = 0; i < 8; ++i) { a[i] = xs[base + i]; a[8 + i] = xs[base + 16 + i]; }
            cq = __builtin_amdgcn_wmma_f32_16x16x32_f16(false, a, false, bq[ks], (short)0, cq, false, false);
            cv = __builtin_amdgcn_wmma_f32_16x16x32_f16(false, a, false, bv[ks], (short)0, cv, false, false);
        }
#pragma unroll
        for (int r = 0; r < 8; ++r) {
            float val = cq[r] * sscale + cv[r];
            int m = mt * 16 + r + half * 8;
            wsh[m * LDST + nglob] = (_Float16)val;
            sumW += val; sumW2 += val * val;
        }
    }
    atomicAdd(&red[0], sumW);
    atomicAdd(&red[1], sumW2);
    __syncthreads();

    // ---- stage 2: G = Wpre @ Wl^T ----
    const float wls = wlsum[nglob];
    v16h bl[4];
#pragma unroll
    for (int ks = 0; ks < 4; ++ks) bl[ks] = frag_from_w(wl16, nglob, ks * 32, klo);

    float sumG = 0.f, sumG2 = 0.f, sumGw = 0.f;
    float* gp = G + ((size_t)b * NN + r0) * DIM;
#pragma unroll
    for (int mt = 0; mt < 8; ++mt) {
        v8f c = {};
#pragma unroll
        for (int ks = 0; ks < 4; ++ks) {
            v16h a;
            const int base = (mt * 16 + nlo) * LDST + ks * 32 + klo;
#pragma unroll
            for (int i = 0; i < 8; ++i) { a[i] = wsh[base + i]; a[8 + i] = wsh[base + 16 + i]; }
            c = __builtin_amdgcn_wmma_f32_16x16x32_f16(false, a, false, bl[ks], (short)0, c, false, false);
        }
#pragma unroll
        for (int r = 0; r < 8; ++r) {
            float g = c[r];
            int m = mt * 16 + r + half * 8;
            gp[(size_t)m * DIM + nglob] = g;
            sumG += g; sumG2 += g * g; sumGw += g * wls;
        }
    }
    atomicAdd(&red[2], sumG);
    atomicAdd(&red[3], sumG2);
    atomicAdd(&red[4], sumGw);
    __syncthreads();
    if (tid < 5) atomicAdd(&st[b * 8 + tid], red[tid]);
}

// ---------------------------------------------------------------------------
// K3: both LayerNorms in closed form from the 5 scalars, then tanh, in place.
// out = tanh(a*G + b[o]),  a = rs1*rs0,  b[o] = -rs1*(rs0*m0*wl_s[o] + m1)
__global__ void __launch_bounds__(256)
k3_tanh(float* G, const float* __restrict__ st, const float* __restrict__ wlsum) {
    const int b = blockIdx.y;
    const float sumW  = st[b * 8 + 0];
    const float sumW2 = st[b * 8 + 1];
    const float sumG  = st[b * 8 + 2];
    const float sumG2 = st[b * 8 + 3];
    const float sumGw = st[b * 8 + 4];
    const float wsum1 = wlsum[DIM];
    const float wsum2 = wlsum[DIM + 1];

    const float inv_cnt = 1.0f / (float)LN_CNT;
    const float m0   = sumW * inv_cnt;
    const float var0 = sumW2 * inv_cnt - m0 * m0;
    const float rs0  = rsqrtf(var0 + EPS);
    const float sumZ  = rs0 * (sumG - m0 * (float)NN * wsum1);
    const float sumZ2 = rs0 * rs0 * (sumG2 - 2.f * m0 * sumGw + m0 * m0 * (float)NN * wsum2);
    const float m1   = sumZ * inv_cnt;
    const float var1 = sumZ2 * inv_cnt - m1 * m1;
    const float rs1  = rsqrtf(var1 + EPS);
    const float a    = rs1 * rs0;
    const float c0   = rs1 * rs0 * m0;   // coefficient of wl_s[o]
    const float c1   = rs1 * m1;

    const int tid = threadIdx.x;
    const int o0 = (tid * 8) & 127;
    size_t base = (size_t)b * LN_CNT + ((size_t)blockIdx.x * 256 + tid) * 8;
    float4* p = (float4*)(G + base);
    const float4* wlp = (const float4*)(wlsum + o0);
    float4 v0 = p[0], v1 = p[1];
    float4 w0 = wlp[0], w1 = wlp[1];
    v0.x = fast_tanh(a * v0.x - c0 * w0.x - c1);
    v0.y = fast_tanh(a * v0.y - c0 * w0.y - c1);
    v0.z = fast_tanh(a * v0.z - c0 * w0.z - c1);
    v0.w = fast_tanh(a * v0.w - c0 * w0.w - c1);
    v1.x = fast_tanh(a * v1.x - c0 * w1.x - c1);
    v1.y = fast_tanh(a * v1.y - c0 * w1.y - c1);
    v1.z = fast_tanh(a * v1.z - c0 * w1.z - c1);
    v1.w = fast_tanh(a * v1.w - c0 * w1.w - c1);
    p[0] = v0; p[1] = v1;
}

// ---------------------------------------------------------------------------
extern "C" void kernel_launch(void* const* d_in, const int* in_sizes, int n_in,
                              void* d_out, int out_size, void* d_ws, size_t ws_size,
                              hipStream_t stream) {
    const float* X  = (const float*)d_in[0];
    const float* Wq = (const float*)d_in[1];
    const float* Wk = (const float*)d_in[2];
    const float* Wv = (const float*)d_in[3];
    const float* Wl = (const float*)d_in[4];
    float* out = (float*)d_out;

    char* ws = (char*)d_ws;
    _Float16* wq16  = (_Float16*)(ws + WS_WQ16);
    _Float16* wk16  = (_Float16*)(ws + WS_WK16);
    _Float16* wv16  = (_Float16*)(ws + WS_WV16);
    _Float16* wl16  = (_Float16*)(ws + WS_WL16);
    float*    wlsum = (float*)   (ws + WS_WLSUM);
    float*    S     = (float*)   (ws + WS_S);
    float*    st    = (float*)   (ws + WS_ST);

    k0_init<<<1, 256, 0, stream>>>(Wq, Wk, Wv, Wl, wq16, wk16, wv16, wl16,
                                   wlsum, S, st);
    dim3 g(NN / RPB, BB);  // (32, 64)
    k1_kv_s <<<g, 256, 0, stream>>>(X, wk16, wv16, S);
    k2_fused<<<g, 256, 0, stream>>>(X, wq16, wv16, wl16, S, wlsum, out, st);
    k3_tanh <<<dim3(LN_CNT / (256 * 8), BB), 256, 0, stream>>>(out, st, wlsum);
}
